// EncoderLayer_gnn_45526653337868
// MI455X (gfx1250) — matrix operand order repaired and checked by reading, not compiled, not run on hardware
//
#include <hip/hip_runtime.h>

typedef _Float16 v2h  __attribute__((ext_vector_type(2)));
typedef _Float16 v8h  __attribute__((ext_vector_type(8)));
typedef _Float16 v16h __attribute__((ext_vector_type(16)));
typedef float    v8f  __attribute__((ext_vector_type(8)));

#define WAVES 2
#define TPB   (WAVES * 32)
#define ACTW  128                 // activation row stride in halves (max layer width)

// fragment base offsets (in 512-half fragments) for each layer's packed weights
#define FRAG_L1 0                 // 64->64  : 2 ksteps * 4 ntiles = 8
#define FRAG_L2 8                 // 64->128 : 2 * 8 = 16
#define FRAG_L3 24                // 128->64 : 4 * 4 = 16
#define FRAG_L4 40                // 64->32  : 2 * 2 = 4
#define NFRAG   44

// ---------------------------------------------------------------------------
// Pack a row-major fp32 weight matrix [K][Nout] into f16 WMMA B-fragments.
// Contiguous-k B 32x16 layout used here: lane = 16*(kk/16) + (n%16), half h -> K = kbase + 16*(lane/16) + h.
// ---------------------------------------------------------------------------
template<int K, int Nout, int FragBase>
__device__ __forceinline__ void pack_w(const float* __restrict__ W, _Float16* s_w, int tid) {
    constexpr int ntiles = Nout / 16;
    constexpr int total  = K * Nout;
    for (int i = tid; i < total; i += TPB) {
        int Kr = i / Nout;
        int Nc = i % Nout;
        int kstep = Kr >> 5, kk = Kr & 31;
        int ntile = Nc >> 4, nn = Nc & 15;
        int lam = ((kk >> 4) << 4) | nn;
        int h   = kk & 15;
        int frag = FragBase + kstep * ntiles + ntile;
        s_w[((frag * 32 + lam) << 4) | h] = (_Float16)W[i];
    }
}

// ---------------------------------------------------------------------------
// Build an A fragment (16x32 f16, MxK) from an LDS activation tile.
// Layout (matches packed B): lane L holds M = L%16; half h -> K = kbase + 16*(L/16) + h.
// => halves 0..15 are 16 contiguous halves (two 16B aligned LDS loads).
// ---------------------------------------------------------------------------
__device__ __forceinline__ v16h load_a_frag(const _Float16* src, int lane, int kbase) {
    int M   = lane & 15;
    int off = (lane >> 4) * 16;
    const v8h* p = (const v8h*)(src + M * ACTW + kbase + off);
    v8h lo = p[0];        // K = kbase+off   .. +7
    v8h hi = p[1];        // K = kbase+off+8 .. +15
    return __builtin_shufflevector(lo, hi, 0,1,2,3,4,5,6,7,8,9,10,11,12,13,14,15);
}

// ---------------------------------------------------------------------------
// One MLP layer on a 16-edge tile: dst = relu(src @ W + b), via WMMA f16->f32.
// C/D layout: vgpr i -> M = i + 8*(lane/16), N = lane%16.
// ReLU done in packed f16 with a single v_pk_max_num_f16 per element pair
// (inline asm: avoids llvm.maxnum's canonicalize max(v,v) preamble).
// ---------------------------------------------------------------------------
template<int KSTEPS, int NTILES>
__device__ __forceinline__ void run_layer(const _Float16* src, _Float16* dst,
                                          const _Float16* s_w, int fragBase,
                                          const float* bias, int lane) {
    const int lamHi = lane >> 4;
    const int col   = lane & 15;
    v16h a[KSTEPS];
#pragma unroll
    for (int ks = 0; ks < KSTEPS; ++ks) a[ks] = load_a_frag(src, lane, ks * 32);
#pragma unroll
    for (int nt = 0; nt < NTILES; ++nt) {
        float bv = bias[nt * 16 + col];
        v8f c;
#pragma unroll
        for (int i = 0; i < 8; ++i) c[i] = bv;
#pragma unroll
        for (int ks = 0; ks < KSTEPS; ++ks) {
            const v16h* bp = (const v16h*)(s_w + (((fragBase + ks * NTILES + nt) * 32 + lane) << 4));
            c = __builtin_amdgcn_wmma_f32_16x16x32_f16(false, a[ks], false, *bp,
                                                       (short)0, c, false, false);
        }
        _Float16* drow = dst + 8 * lamHi * ACTW + nt * 16 + col;
#pragma unroll
        for (int i = 0; i < 4; ++i) {
            v2h p;
            p.x = (_Float16)c[2 * i];         // v_cvt_pk_f16_f32 pair
            p.y = (_Float16)c[2 * i + 1];
            asm("v_pk_max_num_f16 %0, %1, 0" : "+v"(p));   // relu both halves, 1 op
            drow[(2 * i)     * ACTW] = p.x;
            drow[(2 * i + 1) * ACTW] = p.y;
        }
    }
}

// ---------------------------------------------------------------------------
// Main edge-MLP kernel. One wave processes 16 edges per tile iteration.
// ---------------------------------------------------------------------------
__global__ void __launch_bounds__(TPB)
edge_mlp_kernel(const float* __restrict__ x, const long long* __restrict__ ei,
                const float* __restrict__ ea,
                const float* __restrict__ W1, const float* __restrict__ b1,
                const float* __restrict__ W2, const float* __restrict__ b2,
                const float* __restrict__ W3, const float* __restrict__ b3,
                const float* __restrict__ W4, const float* __restrict__ b4,
                float* __restrict__ sums, float* __restrict__ cnt, int E)
{
    alignas(32) __shared__ _Float16 s_w[NFRAG * 512];            // 45,056 B packed weights
    __shared__ float                s_bias[288];                 // b1|b2|b3|b4
    alignas(32) __shared__ _Float16 s_act[WAVES * 2 * 16 * ACTW];// per-wave ping/pong tiles
    __shared__ int                  s_dst[WAVES * 16];

    const int tid = threadIdx.x;

    pack_w< 64,  64, FRAG_L1>(W1, s_w, tid);
    pack_w< 64, 128, FRAG_L2>(W2, s_w, tid);
    pack_w<128,  64, FRAG_L3>(W3, s_w, tid);
    pack_w< 64,  32, FRAG_L4>(W4, s_w, tid);
    for (int i = tid; i < 288; i += TPB) {
        float v;
        if      (i <  64) v = b1[i];
        else if (i < 192) v = b2[i - 64];
        else if (i < 256) v = b3[i - 192];
        else              v = b4[i - 256];
        s_bias[i] = v;
    }
    __syncthreads();

    const int wave  = tid >> 5;
    const int lane  = tid & 31;
    const int lamHi = lane >> 4;
    const int col   = lane & 15;
    const int row   = lane & 15;

    _Float16* act0 = s_act + wave * (2 * 16 * ACTW);
    _Float16* act1 = act0 + 16 * ACTW;
    int*      dstW = s_dst + wave * 16;

    const long long* dstp = ei + E;   // edge_index[1] (targets), int64
    const int tiles  = E >> 4;
    const int gw     = blockIdx.x * WAVES + wave;
    const int stride = gridDim.x * WAVES;

    for (int t = gw; t < tiles; t += stride) {
        const int e0 = t << 4;

        int node = (int)dstp[e0 + row];
        if (lane < 16) dstW[lane] = node;

        // Stage f16 input tile: cols 0-31 = x[dst] (lanes 0-15), 32-63 = edge_attr (lanes 16-31)
        const float4* rp = (lane < 16)
            ? (const float4*)(x  + (size_t)node * 32)
            : (const float4*)(ea + (size_t)(e0 + row) * 32);
        _Float16* arow = act0 + row * ACTW + lamHi * 32;
#pragma unroll
        for (int c4 = 0; c4 < 8; ++c4) {
            float4 v = rp[c4];
            arow[c4 * 4 + 0] = (_Float16)v.x;
            arow[c4 * 4 + 1] = (_Float16)v.y;
            arow[c4 * 4 + 2] = (_Float16)v.z;
            arow[c4 * 4 + 3] = (_Float16)v.w;
        }

        run_layer<2, 4>(act0, act1, s_w, FRAG_L1, s_bias + 0,   lane);
        run_layer<2, 8>(act1, act0, s_w, FRAG_L2, s_bias + 64,  lane);
        run_layer<4, 4>(act0, act1, s_w, FRAG_L3, s_bias + 192, lane);

        // Final layer 64->32: WMMA then relu + scatter-add to target nodes
        {
            v16h a0 = load_a_frag(act1, lane, 0);
            v16h a1 = load_a_frag(act1, lane, 32);
#pragma unroll
            for (int nt = 0; nt < 2; ++nt) {
                float bv = s_bias[256 + nt * 16 + col];
                v8f c;
#pragma unroll
                for (int i = 0; i < 8; ++i) c[i] = bv;
                const v16h* bp0 = (const v16h*)(s_w + (((FRAG_L4 + 0 * 2 + nt) * 32 + lane) << 4));
                const v16h* bp1 = (const v16h*)(s_w + (((FRAG_L4 + 1 * 2 + nt) * 32 + lane) << 4));
                c = __builtin_amdgcn_wmma_f32_16x16x32_f16(false, a0, false, *bp0, (short)0, c, false, false);
                c = __builtin_amdgcn_wmma_f32_16x16x32_f16(false, a1, false, *bp1, (short)0, c, false, false);
#pragma unroll
                for (int i = 0; i < 8; ++i) {
                    int m = i + 8 * lamHi;
                    float v = c[i];
                    asm("v_max_num_f32 %0, %1, 0" : "+v"(v));   // single-op relu
                    unsafeAtomicAdd(sums + (size_t)dstW[m] * 32 + nt * 16 + col, v);
                }
            }
        }
        if (lane < 16) unsafeAtomicAdd(cnt + dstW[lane], 1.0f);
    }
}

__global__ void zero_kernel(float* out, float* cnt, int nOut, int nCnt) {
    int i = blockIdx.x * blockDim.x + threadIdx.x;
    if (i < nOut) out[i] = 0.f;
    if (i < nCnt) cnt[i] = 0.f;
}

__global__ void norm_kernel(float* out, const float* __restrict__ cnt, int N) {
    int i = blockIdx.x * blockDim.x + threadIdx.x;
    if (i < N * 32) {
        float c = cnt[i >> 5];
        out[i] /= (c > 1.f ? c : 1.f);
    }
}

extern "C" void kernel_launch(void* const* d_in, const int* in_sizes, int n_in,
                              void* d_out, int out_size, void* d_ws, size_t ws_size,
                              hipStream_t stream) {
    const float*     x  = (const float*)d_in[0];
    const long long* ei = (const long long*)d_in[1];     // int64 [2, E]
    const float*     ea = (const float*)d_in[2];
    const float* W1 = (const float*)d_in[3];  const float* b1 = (const float*)d_in[4];
    const float* W2 = (const float*)d_in[5];  const float* b2 = (const float*)d_in[6];
    const float* W3 = (const float*)d_in[7];  const float* b3 = (const float*)d_in[8];
    const float* W4 = (const float*)d_in[9];  const float* b4 = (const float*)d_in[10];
    float* out = (float*)d_out;

    const int N = in_sizes[0] / 32;
    const int E = in_sizes[2] / 32;
    float* cnt = (float*)d_ws;     // N floats of scratch for per-node counts

    zero_kernel<<<(N * 32 + 255) / 256, 256, 0, stream>>>(out, cnt, N * 32, N);

    int tiles  = E >> 4;
    int blocks = (tiles + WAVES - 1) / WAVES;
    if (blocks > 2048) blocks = 2048;
    if (blocks < 1)    blocks = 1;
    edge_mlp_kernel<<<blocks, TPB, 0, stream>>>(x, ei, ea, W1, b1, W2, b2, W3, b3, W4, b4,
                                                out, cnt, E);

    norm_kernel<<<(N * 32 + 255) / 256, 256, 0, stream>>>(out, cnt, N);
}
